// PointerTransformer_78786880077913
// MI455X (gfx1250) — compile-verified, matrix operand order checked by reading, not run on hardware
//
#include <hip/hip_runtime.h>
#include <hip/hip_bf16.h>
#include <math.h>

#define B_SZ   32
#define N_TOK  256
#define D_MOD  256
#define N_HEAD 8
#define H_DIM  32
#define D_FF   1024

typedef _Float16 h16;
typedef __attribute__((ext_vector_type(16))) _Float16 v16h;
typedef __attribute__((ext_vector_type(8)))  float    v8f;

// ============================================================================
// Weight pre-pack: f32 (K,N) row-major  ->  f16 fragment-major tiles.
// Tile (kt,nt) covers K rows [kt*32,kt*32+32), N cols [nt*16,nt*16+16).
// Within a tile: lane (0..31) * 16 contiguous halfs, laid out per the
// ISA 7.12.2 B-matrix fragment: n = nt*16 + (lane&15), hh = lane>>4,
// element j -> k = kt*32 + (j>>3)*16 + hh*8 + (j&7).
// A B-fragment load is then exactly two global_load_b128 per lane.
// ============================================================================
__global__ void k_pack(const float* __restrict__ Wsrc, h16* __restrict__ Wdst,
                       int K, int Nn) {
  const long id = (long)blockIdx.x * 256 + threadIdx.x;   // one thread per half
  const long tile   = id >> 9;                            // kt * (Nn/16) + nt
  const int  within = (int)(id & 511);                    // lane*16 + j
  const int  lane = within >> 4, j = within & 15;
  const int  ntn  = Nn >> 4;
  const int  kt = (int)(tile / ntn), nt = (int)(tile % ntn);
  const int  k = (kt << 5) + ((j >> 3) << 4) + ((lane >> 4) << 3) + (j & 7);
  const int  n = (nt << 4) + (lane & 15);
  Wdst[id] = (h16)Wsrc[(long)k * Nn + n];
}

// ---------------- A-fragment loader (f32 row-major -> v16h) -----------------
__device__ inline v16h load_frag_a(const float* __restrict__ row, int hh8) {
  v16h a;
#pragma unroll
  for (int j = 0; j < 8; ++j) {
    a[j]     = (h16)row[hh8 + j];
    a[j + 8] = (h16)row[hh8 + 16 + j];
  }
  return a;
}

// ============================================================================
// GEMM (packed weights):  C = act(A @ Bp + bias)
// A: (M,K) f32 rm activations ; Bp: packed f16 tiles ; C: (M,N) f32.
// One wave per 16x64 output strip (4 n-tiles share one A fragment),
// 4 waves/block, K unrolled x2 (K % 64 == 0) -> 8 WMMAs per iteration.
// ============================================================================
__global__ void k_gemm_p(const float* __restrict__ A, const h16* __restrict__ Bp,
                         const float* __restrict__ bias, float* __restrict__ C,
                         int K, int Nn, int relu) {
  const int lane = threadIdx.x;
  const int mt   = blockIdx.y * blockDim.y + threadIdx.y;   // m-tile
  const int ntg  = blockIdx.x;                              // group of 4 n-tiles
  const int hh8  = (lane >> 4) << 3;
  const float* Arow = A + (long)(mt * 16 + (lane & 15)) * K;
  const long  bstep = (long)(Nn >> 4) * 512;                // one kt step
  const h16* bq[4];
#pragma unroll
  for (int q = 0; q < 4; ++q)
    bq[q] = Bp + ((long)(ntg * 4 + q) * 32 + lane) * 16;    // kt = 0 tiles

  v8f acc[4] = {};
  for (int k0 = 0; k0 < K; k0 += 64) {
    __builtin_prefetch(Arow + k0 + 64, 0, 1);               // global_prefetch_b8
#pragma unroll
    for (int u = 0; u < 2; ++u) {
      v16h a = load_frag_a(Arow + k0 + u * 32, hh8);
#pragma unroll
      for (int q = 0; q < 4; ++q) {
        v16h b = *(const v16h*)bq[q];
        bq[q] += bstep;
        acc[q] = __builtin_amdgcn_wmma_f32_16x16x32_f16(
            false, a, false, b, (short)0, acc[q], false, false);
      }
    }
  }

  const int mbase = mt * 16 + hh8;
#pragma unroll
  for (int q = 0; q < 4; ++q) {
    const int n    = (ntg * 4 + q) * 16 + (lane & 15);
    const float bv = bias ? bias[n] : 0.0f;
#pragma unroll
    for (int r = 0; r < 8; ++r) {
      float v = acc[q][r] + bv;
      if (relu) v = v > 0.0f ? v : 0.0f;
      C[(long)(mbase + r) * Nn + n] = v;
    }
  }
}

// ============================================================================
// Batched GEMM, B transposed:  C[bz] = scale * (A[bz] @ Bt[bz]^T)
// A: (M,K) f32 rm ; Bt: (N,K) f32 rm -> both contiguous along K.
// Same 16x64-per-wave structure. Used for q @ keys^T / sqrt(D).
// ============================================================================
__global__ void k_gemm_t(const float* __restrict__ A, const float* __restrict__ Bt,
                         float* __restrict__ C, int K, int Nn,
                         long sA, long sB, long sC, float scale) {
  const int lane = threadIdx.x;
  const int mt   = blockIdx.y * blockDim.y + threadIdx.y;
  const int ntg  = blockIdx.x;
  const int bz   = blockIdx.z;
  const int hh8  = (lane >> 4) << 3;
  const float* Arow = A + bz * sA + (long)(mt * 16 + (lane & 15)) * K;
  const float* br[4];
#pragma unroll
  for (int q = 0; q < 4; ++q)
    br[q] = Bt + bz * sB + (long)((ntg * 4 + q) * 16 + (lane & 15)) * K;
  float* Cb = C + bz * sC;

  v8f acc[4] = {};
  for (int k0 = 0; k0 < K; k0 += 64) {
#pragma unroll
    for (int u = 0; u < 2; ++u) {
      v16h a = load_frag_a(Arow + k0 + u * 32, hh8);
#pragma unroll
      for (int q = 0; q < 4; ++q) {
        v16h b = load_frag_a(br[q] + k0 + u * 32, hh8);   // same fragment pattern
        acc[q] = __builtin_amdgcn_wmma_f32_16x16x32_f16(
            false, a, false, b, (short)0, acc[q], false, false);
      }
    }
  }

  const int mbase = mt * 16 + hh8;
#pragma unroll
  for (int q = 0; q < 4; ++q) {
    const int n = (ntg * 4 + q) * 16 + (lane & 15);
#pragma unroll
    for (int r = 0; r < 8; ++r)
      Cb[(long)(mbase + r) * Nn + n] = acc[q][r] * scale;
  }
}

// ---------------- Input projection + positional encoding -------------------
__global__ void k_embed(const float* __restrict__ x, const float* __restrict__ w,
                        const float* __restrict__ b, float* __restrict__ H) {
  const int row = blockIdx.x;           // b*N + n
  const int d   = threadIdx.x;          // 0..255
  const int n   = row & (N_TOK - 1);
  const float* xr = x + (long)row * 16;
  float acc = b[d];
#pragma unroll
  for (int k = 0; k < 16; ++k) acc += xr[k] * w[k * D_MOD + d];
  const int i2 = d & ~1;
  const float div = __expf(-(logf(10000.0f) / (float)D_MOD) * (float)i2);
  const float ang = (float)n * div;
  acc += (d & 1) ? cosf(ang) : sinf(ang);
  H[(long)row * D_MOD + d] = acc;
}

// ---------------- LayerNorm with residual: out = LN(xin + res)*g + b -------
__global__ void k_ln(float* __restrict__ out, const float* __restrict__ xin,
                     const float* __restrict__ res, const float* __restrict__ g,
                     const float* __restrict__ bvec) {
  __shared__ float red[D_MOD];
  const int row = blockIdx.x, d = threadIdx.x;
  const long off = (long)row * D_MOD + d;
  const float v = xin[off] + res[off];
  red[d] = v; __syncthreads();
  for (int s = 128; s > 0; s >>= 1) { if (d < s) red[d] += red[d + s]; __syncthreads(); }
  const float mean = red[0] * (1.0f / D_MOD); __syncthreads();
  const float c = v - mean;
  red[d] = c * c; __syncthreads();
  for (int s = 128; s > 0; s >>= 1) { if (d < s) red[d] += red[d + s]; __syncthreads(); }
  const float var = red[0] * (1.0f / D_MOD);
  out[off] = c * rsqrtf(var + 1e-5f) * g[d] + bvec[d];
}

// ---------------- Attention core: one block per (b, head, q-row) -----------
__global__ void k_attn(const float* __restrict__ Q, const float* __restrict__ Kb,
                       const float* __restrict__ Vb, float* __restrict__ O, int causal) {
  __shared__ float sc[N_TOK];
  __shared__ float red[N_TOK];
  __shared__ float qs[H_DIM];
  const int b = blockIdx.z, hh = blockIdx.y, t = blockIdx.x;
  const int j = threadIdx.x;
  const long qoff = ((long)(b * N_TOK + t)) * D_MOD + hh * H_DIM;
  if (j < H_DIM) qs[j] = Q[qoff + j];
  __syncthreads();

  const float* krow = Kb + ((long)(b * N_TOK + j)) * D_MOD + hh * H_DIM;
  float s = 0.0f;
#pragma unroll
  for (int d2 = 0; d2 < H_DIM; ++d2) s += qs[d2] * krow[d2];
  s *= 0.17677669529663687f;            // 1/sqrt(32)
  if (causal && j > t) s = -1e9f;

  sc[j] = s; red[j] = s; __syncthreads();
  for (int st = 128; st > 0; st >>= 1) { if (j < st) red[j] = fmaxf(red[j], red[j + st]); __syncthreads(); }
  const float mx = red[0]; __syncthreads();
  const float e = __expf(s - mx);
  sc[j] = e; red[j] = e; __syncthreads();
  for (int st = 128; st > 0; st >>= 1) { if (j < st) red[j] += red[j + st]; __syncthreads(); }
  const float Z = red[0]; __syncthreads();

  if (j < H_DIM) {
    float acc = 0.0f;
    for (int kk = 0; kk < N_TOK; ++kk)
      acc += sc[kk] * Vb[((long)(b * N_TOK + kk)) * D_MOD + hh * H_DIM + j];
    O[qoff + j] = acc / Z;
  }
}

// ---------------- Decoder input: [query_start, H[y_0..y_{N-2}]] ------------
__global__ void k_build_tgt(const float* __restrict__ H, const int* __restrict__ idx,
                            const float* __restrict__ qstart, float* __restrict__ Dc) {
  const int b = blockIdx.y, t = blockIdx.x, d = threadIdx.x;
  float v;
  if (t == 0) v = qstart[d];
  else {
    const int y = idx[b * N_TOK + t - 1];
    v = H[((long)(b * N_TOK + y)) * D_MOD + d];
  }
  Dc[((long)(b * N_TOK + t)) * D_MOD + d] = v;
}

// ---------------- Permutation inverse: pos[b, y_t] = t ----------------------
__global__ void k_pos(const int* __restrict__ idx, int* __restrict__ pos) {
  const int i = blockIdx.x * blockDim.x + threadIdx.x;  // b*N + t
  const int b = i / N_TOK, t = i % N_TOK;
  pos[b * N_TOK + idx[i]] = t;
}

// ---------------- Edge-bias MLP on gathered rows only -----------------------
__global__ void k_edge(const float* __restrict__ ef, const int* __restrict__ idx,
                       const float* __restrict__ w1, const float* __restrict__ b1,
                       const float* __restrict__ w2, const float* __restrict__ b2,
                       float* __restrict__ bias) {
  const int b = blockIdx.y, t = blockIdx.x, j = threadIdx.x;
  float v = 0.0f;
  if (t > 0) {
    const int i = idx[b * N_TOK + t - 1];
    const float* e = ef + (((long)(b * N_TOK) + i) * N_TOK + j) * 3;
    const float e0 = e[0], e1 = e[1], e2 = e[2];
    float acc = b2[0];
#pragma unroll
    for (int kk = 0; kk < 32; ++kk) {
      float hv = e0 * w1[kk] + e1 * w1[32 + kk] + e2 * w1[64 + kk] + b1[kk];
      hv = hv > 0.0f ? hv : 0.0f;
      acc += hv * w2[kk];
    }
    v = fminf(fmaxf(acc, -10.0f), 10.0f);
  }
  bias[(((long)b * N_TOK) + t) * N_TOK + j] = v;
}

// ---------------- Final: bias + clip + visited mask + log_softmax + NLL -----
__global__ void k_nll(const float* __restrict__ logits, const float* __restrict__ bias,
                      const int* __restrict__ pos, const int* __restrict__ idx,
                      float* __restrict__ out) {
  __shared__ float red[N_TOK];
  const int b = blockIdx.y, t = blockIdx.x, n = threadIdx.x;
  const long base = (((long)b * N_TOK) + t) * N_TOK;
  float l = logits[base + n] + bias[base + n];
  l = fminf(fmaxf(l, -50.0f), 50.0f);
  if (pos[b * N_TOK + n] < t) l = -10000.0f;
  red[n] = l; __syncthreads();
  for (int s = 128; s > 0; s >>= 1) { if (n < s) red[n] = fmaxf(red[n], red[n + s]); __syncthreads(); }
  const float mx = red[0]; __syncthreads();
  red[n] = __expf(l - mx); __syncthreads();
  for (int s = 128; s > 0; s >>= 1) { if (n < s) red[n] += red[n + s]; __syncthreads(); }
  const float lse = mx + __logf(red[0]);
  if (n == 0) {
    const int y = idx[b * N_TOK + t];
    float ly = logits[base + y] + bias[base + y];
    ly = fminf(fmaxf(ly, -50.0f), 50.0f);
    atomicAdd(out, (lse - ly) * (1.0f / (float)(B_SZ * N_TOK)));
  }
}

__global__ void k_zero(float* p) { p[0] = 0.0f; }

// ============================================================================
extern "C" void kernel_launch(void* const* d_in, const int* in_sizes, int n_in,
                              void* d_out, int out_size, void* d_ws, size_t ws_size,
                              hipStream_t stream) {
  (void)in_sizes; (void)n_in; (void)out_size; (void)ws_size;
  int ix = 0;
  auto Pp = [&]() { return (const float*)d_in[ix++]; };

  const float* x        = Pp();                      // x (B,N,16)
  const int*   tgt_idx  = (const int*)d_in[ix++];    // target_idx (B,N)
  const float* edge     = Pp();                      // edge_feats (B,N,N,3)
  const float* inp_w    = Pp();
  const float* inp_b    = Pp();

  // enc layers: sa{wq,bq,wk,bk,wv,bv,wo,bo}, ff{w1,b1,w2,b2}, ln1g,ln1b,ln2g,ln2b
  const float* enc_p[2][16];
  for (int l = 0; l < 2; ++l)
    for (int i = 0; i < 16; ++i) enc_p[l][i] = Pp();
  // dec layers: sa{8}, ca{8}, ff{4}, ln1g,ln1b,ln2g,ln2b,ln3g,ln3b
  const float* dec_p[2][26];
  for (int l = 0; l < 2; ++l)
    for (int i = 0; i < 26; ++i) dec_p[l][i] = Pp();

  const float* query_start = Pp();
  const float* out_w = Pp();  const float* out_b = Pp();
  const float* ptr_w = Pp();  const float* ptr_b = Pp();
  const float* eb_w1 = Pp();  const float* eb_b1 = Pp();
  const float* eb_w2 = Pp();  const float* eb_b2 = Pp();

  // ---- workspace layout (float units) ----
  const long SZ = (long)B_SZ * N_TOK * D_MOD;   // 2,097,152
  float* W    = (float*)d_ws;
  float* H    = W;
  float* Dc   = W + 1 * SZ;
  float* TMP  = W + 2 * SZ;
  float* Qb   = W + 3 * SZ;
  float* Kc   = W + 4 * SZ;
  float* Vb   = W + 5 * SZ;
  float* Ob   = W + 6 * SZ;
  float* FF   = W + 7 * SZ;                      // B*N*DFF = 4*SZ
  float* BIAS = W + 11 * SZ;
  float* LG   = W + 12 * SZ;
  int*   POS  = (int*)(W + 13 * SZ);             // 8192 ints
  h16*   packCur = (h16*)(W + 13 * SZ + 8192);   // ~7.6 MB of packed f16

  const int M = B_SZ * N_TOK;                    // 8192 rows

  // ---- pack all weight matrices to f16 fragment-major ----
  auto pack = [&](const float* w, int K, int Nn) -> const h16* {
    h16* dst = packCur;
    const long total = (long)K * Nn;
    k_pack<<<(unsigned)(total / 256), 256, 0, stream>>>(w, dst, K, Nn);
    packCur += total;
    return dst;
  };
  const h16* pe[2][6];   // wq,wk,wv,wo,w1,w2
  for (int l = 0; l < 2; ++l) {
    pe[l][0] = pack(enc_p[l][0], D_MOD, D_MOD);
    pe[l][1] = pack(enc_p[l][2], D_MOD, D_MOD);
    pe[l][2] = pack(enc_p[l][4], D_MOD, D_MOD);
    pe[l][3] = pack(enc_p[l][6], D_MOD, D_MOD);
    pe[l][4] = pack(enc_p[l][8], D_MOD, D_FF);
    pe[l][5] = pack(enc_p[l][10], D_FF, D_MOD);
  }
  const h16* pd[2][10];  // sa wq,wk,wv,wo | ca wq,wk,wv,wo | w1,w2
  for (int l = 0; l < 2; ++l) {
    pd[l][0] = pack(dec_p[l][0],  D_MOD, D_MOD);
    pd[l][1] = pack(dec_p[l][2],  D_MOD, D_MOD);
    pd[l][2] = pack(dec_p[l][4],  D_MOD, D_MOD);
    pd[l][3] = pack(dec_p[l][6],  D_MOD, D_MOD);
    pd[l][4] = pack(dec_p[l][8],  D_MOD, D_MOD);
    pd[l][5] = pack(dec_p[l][10], D_MOD, D_MOD);
    pd[l][6] = pack(dec_p[l][12], D_MOD, D_MOD);
    pd[l][7] = pack(dec_p[l][14], D_MOD, D_MOD);
    pd[l][8] = pack(dec_p[l][16], D_MOD, D_FF);
    pd[l][9] = pack(dec_p[l][18], D_FF, D_MOD);
  }
  const h16* p_outw = pack(out_w, D_MOD, D_MOD);
  const h16* p_ptrw = pack(ptr_w, D_MOD, D_MOD);

  auto gemm = [&](const float* A, const h16* Bp, const float* bias, float* C,
                  int Mm, int K, int Nn, int relu) {
    dim3 grid(Nn / 64, Mm / 64, 1);
    dim3 block(32, 4, 1);
    k_gemm_p<<<grid, block, 0, stream>>>(A, Bp, bias, C, K, Nn, relu);
  };
  auto ln = [&](float* out, const float* xin, const float* res,
                const float* g, const float* bvec) {
    k_ln<<<M, D_MOD, 0, stream>>>(out, xin, res, g, bvec);
  };
  auto attn = [&](const float* Q, const float* K, const float* V, float* O, int causal) {
    k_attn<<<dim3(N_TOK, N_HEAD, B_SZ), N_TOK, 0, stream>>>(Q, K, V, O, causal);
  };

  k_zero<<<1, 1, 0, stream>>>((float*)d_out);

  // ---- encoder ----
  k_embed<<<M, D_MOD, 0, stream>>>(x, inp_w, inp_b, H);
  for (int l = 0; l < 2; ++l) {
    const float* const* p = enc_p[l];
    gemm(H, pe[l][0], p[1], Qb, M, D_MOD, D_MOD, 0);    // wq
    gemm(H, pe[l][1], p[3], Kc, M, D_MOD, D_MOD, 0);    // wk
    gemm(H, pe[l][2], p[5], Vb, M, D_MOD, D_MOD, 0);    // wv
    attn(Qb, Kc, Vb, Ob, /*causal=*/0);
    gemm(Ob, pe[l][3], p[7], TMP, M, D_MOD, D_MOD, 0);  // wo
    ln(H, TMP, H, p[12], p[13]);                        // ln1
    gemm(H, pe[l][4], p[9], FF, M, D_MOD, D_FF, 1);     // w1 + relu
    gemm(FF, pe[l][5], p[11], TMP, M, D_FF, D_MOD, 0);  // w2
    ln(H, TMP, H, p[14], p[15]);                        // ln2
  }

  // ---- decoder (single causal pass over teacher-forced tgt) ----
  k_build_tgt<<<dim3(N_TOK, B_SZ), D_MOD, 0, stream>>>(H, tgt_idx, query_start, Dc);
  for (int l = 0; l < 2; ++l) {
    const float* const* p = dec_p[l];
    // self-attn (causal)
    gemm(Dc, pd[l][0], p[1], Qb, M, D_MOD, D_MOD, 0);
    gemm(Dc, pd[l][1], p[3], Kc, M, D_MOD, D_MOD, 0);
    gemm(Dc, pd[l][2], p[5], Vb, M, D_MOD, D_MOD, 0);
    attn(Qb, Kc, Vb, Ob, /*causal=*/1);
    gemm(Ob, pd[l][3], p[7], TMP, M, D_MOD, D_MOD, 0);
    ln(Dc, TMP, Dc, p[20], p[21]);                      // ln1
    // cross-attn vs encoder memory H
    gemm(Dc, pd[l][4], p[9],  Qb, M, D_MOD, D_MOD, 0);
    gemm(H,  pd[l][5], p[11], Kc, M, D_MOD, D_MOD, 0);
    gemm(H,  pd[l][6], p[13], Vb, M, D_MOD, D_MOD, 0);
    attn(Qb, Kc, Vb, Ob, /*causal=*/0);
    gemm(Ob, pd[l][7], p[15], TMP, M, D_MOD, D_MOD, 0);
    ln(Dc, TMP, Dc, p[22], p[23]);                      // ln2
    // ffn
    gemm(Dc, pd[l][8], p[17], FF, M, D_MOD, D_FF, 1);
    gemm(FF, pd[l][9], p[19], TMP, M, D_FF, D_MOD, 0);
    ln(Dc, TMP, Dc, p[24], p[25]);                      // ln3
  }

  // ---- pointer logits: q = Dc@out_w+b ; keys = H@ptr_w+b ; q keys^T / 16
  gemm(Dc, p_outw, out_b, Qb, M, D_MOD, D_MOD, 0);
  gemm(H,  p_ptrw, ptr_b, Kc, M, D_MOD, D_MOD, 0);
  {
    dim3 grid(N_TOK / 64, N_TOK / 64, B_SZ);
    dim3 block(32, 4, 1);
    k_gemm_t<<<grid, block, 0, stream>>>(Qb, Kc, LG, D_MOD, N_TOK,
                                         (long)N_TOK * D_MOD, (long)N_TOK * D_MOD,
                                         (long)N_TOK * N_TOK, 0.0625f);
  }

  // ---- edge bias, visited mask, NLL ----
  k_pos<<<B_SZ, N_TOK, 0, stream>>>(tgt_idx, POS);
  k_edge<<<dim3(N_TOK, B_SZ), N_TOK, 0, stream>>>(edge, tgt_idx, eb_w1, eb_b1,
                                                  eb_w2, eb_b2, BIAS);
  k_nll<<<dim3(N_TOK, B_SZ), N_TOK, 0, stream>>>(LG, BIAS, POS, tgt_idx,
                                                 (float*)d_out);
}
